// GraphConv_35837207118175
// MI455X (gfx1250) — matrix operand order repaired and checked by reading, not compiled, not run on hardware
//
#include <hip/hip_runtime.h>

#define NUSR 40000
#define NITM 25000
#define NENT 100000
#define NEDGE 600000
#define NINTER 600000
#define DIMV 64
#define GAMMA_C 0.05f

typedef __attribute__((ext_vector_type(16))) _Float16 v16h;
typedef __attribute__((ext_vector_type(8)))  float    v8f;

__device__ __forceinline__ float sigm_(float x) { return 1.0f / (1.0f + expf(-x)); }

// ---------------- utility kernels ----------------

__global__ void kzero(float* __restrict__ p, long n) {
    long i = (long)blockIdx.x * blockDim.x + threadIdx.x;
    if (i < n) p[i] = 0.0f;
}

// a[i] /= max(cnt[i/64], 1)
__global__ void kmean(float* __restrict__ a, const float* __restrict__ cnt, long n) {
    long i = (long)blockIdx.x * blockDim.x + threadIdx.x;
    if (i < n) a[i] = a[i] / fmaxf(cnt[i >> 6], 1.0f);
}

// ---------------- KG edge scatter (segment sums) ----------------

__global__ void __launch_bounds__(256) kedge_scatter(
    const int* __restrict__ ei, const int* __restrict__ et,
    const int* __restrict__ drop, int hop,
    const float* __restrict__ ent, const float* __restrict__ relw,
    float* __restrict__ agg1s, float* __restrict__ agg2s, float* __restrict__ rels,
    float* __restrict__ cwc, float* __restrict__ cws, float* __restrict__ cew)
{
    long tid = (long)blockIdx.x * blockDim.x + threadIdx.x;
    if (tid >= (long)NEDGE * 16) return;
    int e = (int)(tid >> 4);
    int c = (int)(tid & 15);
    int m = drop[e];
    for (int h = 1; h <= hop; ++h) m &= drop[h * NEDGE + e];
    if (!m) return;
    int head = ei[e];
    int tail = ei[NEDGE + e];
    bool cross = (head < NITM) != (tail < NITM);
    int ty = et[e];
    int d0 = c * 4;
    const float4 t4 = *(const float4*)(ent + (size_t)tail * DIMV + d0);
    const float4 r4 = *(const float4*)(relw + (size_t)ty * DIMV + d0);
    size_t ho = (size_t)head * DIMV + d0;
    if (cross) {
        atomicAdd(agg1s + ho + 0, t4.x * r4.x);
        atomicAdd(agg1s + ho + 1, t4.y * r4.y);
        atomicAdd(agg1s + ho + 2, t4.z * r4.z);
        atomicAdd(agg1s + ho + 3, t4.w * r4.w);
    } else {
        atomicAdd(agg2s + ho + 0, t4.x + r4.x);
        atomicAdd(agg2s + ho + 1, t4.y + r4.y);
        atomicAdd(agg2s + ho + 2, t4.z + r4.z);
        atomicAdd(agg2s + ho + 3, t4.w + r4.w);
    }
    atomicAdd(rels + ho + 0, r4.x);
    atomicAdd(rels + ho + 1, r4.y);
    atomicAdd(rels + ho + 2, r4.z);
    atomicAdd(rels + ho + 3, r4.w);
    if (c == 0) {
        if (cross) atomicAdd(cwc + head, 1.0f);
        else       atomicAdd(cws + head, 1.0f);
        atomicAdd(cew + head, 1.0f);
    }
}

// ---------------- fused dual GEMM with WMMA + async global->LDS staging ----------------
// ent_new = leaky(A1 @ W1^T + b1)/2 + leaky(A2 @ W2^T + b2)/2
// One wave = one 16-row tile; 4 N-tiles of 16 cols; K=64 as two k-chunks of 32.
// A tiles (16x64 f32 = 4KB contiguous) are staged into LDS with
// global_load_async_to_lds_b128, tracked by ASYNCcnt.

// Issue a 4KB contiguous tile load (row-clamped) from global into LDS.
__device__ __forceinline__ void async_tile_load(const float* __restrict__ src,
                                                unsigned lds_base,
                                                int row_base, int nrows, int lane)
{
#pragma unroll
    for (int it = 0; it < 8; ++it) {
        int ci = it * 32 + lane;                    // 256 chunks of 16B
        int ro = ci >> 4;                           // row within tile
        int bo = (ci & 15) * 16;                    // byte within row
        int r  = row_base + ro;
        if (r > nrows - 1) r = nrows - 1;           // clamp (dup row, outputs predicated)
        unsigned long long ga = (unsigned long long)(const void*)src
                              + (unsigned long long)r * 256ull + (unsigned)bo;
        unsigned la = lds_base + (unsigned)(ci * 16);
        asm volatile("global_load_async_to_lds_b128 %0, %1, off"
                     :: "v"(la), "v"(ga)
                     : "memory");
    }
}

__global__ void __launch_bounds__(256) kgemm_fused(
    const float* __restrict__ A1, const float* __restrict__ A2,
    const float* __restrict__ W1, const float* __restrict__ B1,
    const float* __restrict__ W2, const float* __restrict__ B2,
    float* __restrict__ out, int nrows)
{
    __shared__ _Float16 sW1[64 * 64];
    __shared__ _Float16 sW2[64 * 64];
    __shared__ float    sAf[8][16 * 64];   // per-wave float staging tile (reused A1 then A2)

    const int tid  = threadIdx.x;
    const int wave = tid >> 5;
    const int lane = tid & 31;
    const int row_base = blockIdx.x * 128 + wave * 16;

    const unsigned ldsA = (unsigned)(size_t)(const void*)&sAf[wave][0];

    // kick off async DMA of the A1 tile; overlaps with W staging below
    async_tile_load(A1, ldsA, row_base, nrows, lane);

    for (int i = tid; i < 64 * 64; i += 256) {
        sW1[i] = (_Float16)W1[i];
        sW2[i] = (_Float16)W2[i];
    }
    __syncthreads();

    const int l  = lane & 15;
    const int hi = lane >> 4;

    // ---- A1 fragments (16x32 f16 layout): v0-3 -> K = hi*8+i, v4-7 -> K = 16+hi*8+(i-8)
    asm volatile("s_wait_asynccnt 0" ::: "memory");
    v16h a1c0, a1c1;
#pragma unroll
    for (int i = 0; i < 16; ++i) {
        int K = (i < 8) ? (hi * 8 + i) : (16 + hi * 8 + (i - 8));
        a1c0[i] = (_Float16)sAf[wave][l * 64 + K];
        a1c1[i] = (_Float16)sAf[wave][l * 64 + K + 32];
    }
    // retire the LDS reads before overwriting the staging buffer with the A2 tile
    asm volatile("s_wait_dscnt 0" ::: "memory");

    async_tile_load(A2, ldsA, row_base, nrows, lane);
    asm volatile("s_wait_asynccnt 0" ::: "memory");
    v16h a2c0, a2c1;
#pragma unroll
    for (int i = 0; i < 16; ++i) {
        int K = (i < 8) ? (hi * 8 + i) : (16 + hi * 8 + (i - 8));
        a2c0[i] = (_Float16)sAf[wave][l * 64 + K];
        a2c1[i] = (_Float16)sAf[wave][l * 64 + K + 32];
    }

    for (int nt = 0; nt < 4; ++nt) {
        int nglob = nt * 16 + l;           // output column; B[k][n] = W[nglob][k]
        v16h b1c0, b1c1, b2c0, b2c1;
#pragma unroll
        for (int i = 0; i < 16; ++i) {
            int K = (i < 8) ? (hi * 8 + i) : (16 + hi * 8 + (i - 8));   // B 32x16 layout: v0-3 -> K = hi*8+i, v4-7 -> K = 16+hi*8+(i-8)
            b1c0[i] = sW1[nglob * 64 + K];
            b1c1[i] = sW1[nglob * 64 + K + 32];
            b2c0[i] = sW2[nglob * 64 + K];
            b2c1[i] = sW2[nglob * 64 + K + 32];
        }
        v8f c1 = {};
        c1 = __builtin_amdgcn_wmma_f32_16x16x32_f16(false, a1c0, false, b1c0, (short)0, c1, false, false);
        c1 = __builtin_amdgcn_wmma_f32_16x16x32_f16(false, a1c1, false, b1c1, (short)0, c1, false, false);
        v8f c2 = {};
        c2 = __builtin_amdgcn_wmma_f32_16x16x32_f16(false, a2c0, false, b2c0, (short)0, c2, false, false);
        c2 = __builtin_amdgcn_wmma_f32_16x16x32_f16(false, a2c1, false, b2c1, (short)0, c2, false, false);

        float bb1 = B1[nglob];
        float bb2 = B2[nglob];
#pragma unroll
        for (int r = 0; r < 8; ++r) {
            int gr = row_base + r + 8 * hi;   // C/D layout: M = r + 8*hi, N = nglob
            if (gr < nrows) {
                float v1 = c1[r] + bb1; v1 = (v1 > 0.0f ? v1 : 0.01f * v1) * 0.5f;
                float v2 = c2[r] + bb2; v2 = (v2 > 0.0f ? v2 : 0.01f * v2) * 0.5f;
                out[(size_t)gr * 64 + nglob] = v1 + v2;
            }
        }
    }
}

// ---------------- interaction-side kernels ----------------

__global__ void __launch_bounds__(256) kitem_scatter(
    const int* __restrict__ im, const float* __restrict__ ucf,
    float* __restrict__ sums, float* __restrict__ cnt)
{
    long tid = (long)blockIdx.x * blockDim.x + threadIdx.x;
    if (tid >= (long)NINTER * 16) return;
    int e = (int)(tid >> 4);
    int c = (int)(tid & 15);
    int row = im[2 * e], col = im[2 * e + 1];
    int d0 = c * 4;
    float4 v = *(const float4*)(ucf + (size_t)row * DIMV + d0);
    size_t co = (size_t)col * DIMV + d0;
    atomicAdd(sums + co + 0, v.x);
    atomicAdd(sums + co + 1, v.y);
    atomicAdd(sums + co + 2, v.z);
    atomicAdd(sums + co + 3, v.w);
    if (c == 0) atomicAdd(cnt + col, 1.0f);
}

__global__ void __launch_bounds__(256) kdots(
    const int* __restrict__ im, const float* __restrict__ u, const float* __restrict__ ucf,
    const float* __restrict__ reli, const float* __restrict__ ikg, const float* __restrict__ icf,
    float* __restrict__ p, float* __restrict__ q, int* __restrict__ mp, int* __restrict__ mq)
{
    int e = blockIdx.x * blockDim.x + threadIdx.x;
    if (e >= NINTER) return;
    int row = im[2 * e], col = im[2 * e + 1];
    const float4* U = (const float4*)(u   + (size_t)row * DIMV);
    const float4* V = (const float4*)(ucf + (size_t)row * DIMV);
    const float4* R = (const float4*)(reli + (size_t)col * DIMV);
    const float4* G = (const float4*)(ikg  + (size_t)col * DIMV);
    const float4* C = (const float4*)(icf  + (size_t)col * DIMV);
    float s1 = 0.0f, s2 = 0.0f;
#pragma unroll 4
    for (int k = 0; k < 16; ++k) {
        float4 a = U[k], r = R[k], g = G[k];
        s1 += a.x * r.x * g.x + a.y * r.y * g.y + a.z * r.z * g.z + a.w * r.w * g.w;
        float4 b = V[k], c4 = C[k];
        s2 += b.x * c4.x + b.y * c4.y + b.z * c4.z + b.w * c4.w;
    }
    float pv = sigm_(s1), qv = sigm_(s2);
    p[e] = pv; q[e] = qv;
    atomicMax(mp + row, __float_as_int(pv));   // pv,qv > 0 -> int compare is float-ordered
    atomicMax(mq + row, __float_as_int(qv));
}

__global__ void __launch_bounds__(256) ksoftexp(
    const int* __restrict__ im, float* __restrict__ p, float* __restrict__ q,
    const int* __restrict__ mp, float* __restrict__ sp,
    const int* __restrict__ mq, float* __restrict__ sq)
{
    int e = blockIdx.x * blockDim.x + threadIdx.x;
    if (e >= NINTER) return;
    int row = im[2 * e];
    float e1 = expf(p[e] - __int_as_float(mp[row]));
    float e2 = expf(q[e] - __int_as_float(mq[row]));
    p[e] = e1; q[e] = e2;
    atomicAdd(sp + row, e1);
    atomicAdd(sq + row, e2);
}

__global__ void __launch_bounds__(256) knormmask(
    const int* __restrict__ im, float* __restrict__ p, float* __restrict__ q,
    const float* __restrict__ sp, const float* __restrict__ sq, float* __restrict__ mask)
{
    int e = blockIdx.x * blockDim.x + threadIdx.x;
    if (e >= NINTER) return;
    int row = im[2 * e];
    float pn = p[e] / sp[row];
    float qn = q[e] / sq[row];
    p[e] = pn; q[e] = qn;
    mask[e] = (fabsf(sigm_(pn) - sigm_(qn)) < GAMMA_C) ? 1.0f : 0.0f;
}

__global__ void __launch_bounds__(256) kscatter_u(
    const int* __restrict__ im, const float* __restrict__ p, const float* __restrict__ q,
    const float* __restrict__ mask, const float* __restrict__ ikg, const float* __restrict__ icf,
    float* __restrict__ un, float* __restrict__ qn)
{
    long tid = (long)blockIdx.x * blockDim.x + threadIdx.x;
    if (tid >= (long)NINTER * 16) return;
    int e = (int)(tid >> 4);
    float mk = mask[e];
    if (mk == 0.0f) return;
    int c = (int)(tid & 15);
    int row = im[2 * e], col = im[2 * e + 1];
    int d0 = c * 4;
    float w1 = p[e] * mk, w2 = q[e] * mk;
    float4 g  = *(const float4*)(ikg + (size_t)col * DIMV + d0);
    float4 cc = *(const float4*)(icf + (size_t)col * DIMV + d0);
    size_t ro = (size_t)row * DIMV + d0;
    atomicAdd(un + ro + 0, g.x * w1);
    atomicAdd(un + ro + 1, g.y * w1);
    atomicAdd(un + ro + 2, g.z * w1);
    atomicAdd(un + ro + 3, g.w * w1);
    atomicAdd(qn + ro + 0, cc.x * w2);
    atomicAdd(qn + ro + 1, cc.y * w2);
    atomicAdd(qn + ro + 2, cc.z * w2);
    atomicAdd(qn + ro + 3, cc.w * w2);
}

// ---------------- row-wise L2 norm kernels (blockDim = 64 = one row) ----------------

__global__ void kl2norm(float* __restrict__ x, int rows) {
    int row = blockIdx.x;
    __shared__ float red[64];
    float v = x[(size_t)row * DIMV + threadIdx.x];
    red[threadIdx.x] = v * v;
    __syncthreads();
    for (int s = 32; s > 0; s >>= 1) {
        if (threadIdx.x < s) red[threadIdx.x] += red[threadIdx.x + s];
        __syncthreads();
    }
    float nrm = fmaxf(sqrtf(red[0]), 1e-12f);
    x[(size_t)row * DIMV + threadIdx.x] = v / nrm;
}

__global__ void knorm_add(const float* __restrict__ src, float* __restrict__ cur,
                          float* __restrict__ res, int rows) {
    int row = blockIdx.x;
    __shared__ float red[64];
    float v = src[(size_t)row * DIMV + threadIdx.x];
    red[threadIdx.x] = v * v;
    __syncthreads();
    for (int s = 32; s > 0; s >>= 1) {
        if (threadIdx.x < s) red[threadIdx.x] += red[threadIdx.x + s];
        __syncthreads();
    }
    float nrm = fmaxf(sqrtf(red[0]), 1e-12f);
    float val = v / nrm;
    size_t i = (size_t)row * DIMV + threadIdx.x;
    cur[i] = val;
    res[i] += val;
}

__global__ void kbank(const float* __restrict__ mask, float* __restrict__ out) {
    int e = blockIdx.x * blockDim.x + threadIdx.x;
    if (e < NINTER) out[e] = mask[e];
}

// ---------------- host launcher ----------------

static inline unsigned div_up(long n, int b) { return (unsigned)((n + b - 1) / b); }

extern "C" void kernel_launch(void* const* d_in, const int* in_sizes, int n_in,
                              void* d_out, int out_size, void* d_ws, size_t ws_size,
                              hipStream_t stream) {
    (void)in_sizes; (void)n_in; (void)out_size; (void)ws_size;

    const float* user_emb   = (const float*)d_in[0];
    const float* entity_emb = (const float*)d_in[1];
    const float* emb_cf     = (const float*)d_in[2];
    const float* relw       = (const float*)d_in[3];
    const float* W1w        = (const float*)d_in[4];
    const float* W1b        = (const float*)d_in[5];
    const float* W2w        = (const float*)d_in[6];
    const float* W2b        = (const float*)d_in[7];
    const int*   ei         = (const int*)d_in[8];
    const int*   et         = (const int*)d_in[9];
    const int*   im         = (const int*)d_in[10];
    const int*   drop       = (const int*)d_in[11];

    const size_t ENT_D = (size_t)NENT * DIMV;
    const size_t USR_D = (size_t)NUSR * DIMV;
    const size_t ITM_D = (size_t)NITM * DIMV;

    float* base = (float*)d_ws;
    size_t o = 0;
    float* ent_cur = base + o; o += ENT_D;
    float* agg1s   = base + o; o += ENT_D;
    float* agg2s   = base + o; o += ENT_D;
    float* rels    = base + o; o += ENT_D;
    float* cwc     = base + o; o += NENT;
    float* cws     = base + o; o += NENT;
    float* cew     = base + o; o += NENT;
    float* ent_new = base + o; o += ENT_D;
    float* usr_cur = base + o; o += USR_D;   // pair (usr_cur, ucf_cur) contiguous
    float* ucf_cur = base + o; o += USR_D;
    float* u_tmp   = base + o; o += USR_D;   // pair (u_tmp, ucf_tmp) contiguous
    float* ucf_tmp = base + o; o += USR_D;
    float* icf_cur = base + o; o += ITM_D;
    float* icf_new = base + o; o += ITM_D;   // (icf_new, cnt_it) contiguous
    float* cnt_it  = base + o; o += NITM;
    float* segm_p  = base + o; o += NUSR;    // (segm_p, segs_p, segm_q, segs_q) contiguous
    float* segs_p  = base + o; o += NUSR;
    float* segm_q  = base + o; o += NUSR;
    float* segs_q  = base + o; o += NUSR;
    float* p_arr   = base + o; o += NINTER;
    float* q_arr   = base + o; o += NINTER;
    float* mask_f  = base + o; o += NINTER;

    float* out_ent  = (float*)d_out;
    float* out_usr  = out_ent + ENT_D;
    float* out_ucf  = out_usr + USR_D;
    float* out_icf  = out_ucf + USR_D;
    float* out_bank = out_icf + ITM_D;

    // initialize residuals and current embeddings from inputs (every call)
    hipMemcpyAsync(ent_cur, entity_emb, ENT_D * 4, hipMemcpyDeviceToDevice, stream);
    hipMemcpyAsync(usr_cur, user_emb,   USR_D * 4, hipMemcpyDeviceToDevice, stream);
    hipMemcpyAsync(ucf_cur, emb_cf,     USR_D * 4, hipMemcpyDeviceToDevice, stream);
    hipMemcpyAsync(icf_cur, emb_cf + USR_D, ITM_D * 4, hipMemcpyDeviceToDevice, stream);
    hipMemcpyAsync(out_ent, entity_emb, ENT_D * 4, hipMemcpyDeviceToDevice, stream);
    hipMemcpyAsync(out_usr, user_emb,   USR_D * 4, hipMemcpyDeviceToDevice, stream);
    hipMemcpyAsync(out_ucf, emb_cf,     USR_D * 4, hipMemcpyDeviceToDevice, stream);
    hipMemcpyAsync(out_icf, emb_cf + USR_D, ITM_D * 4, hipMemcpyDeviceToDevice, stream);

    for (int hop = 0; hop < 2; ++hop) {
        // KG forward: weighted segment means over heads
        long zn = (long)(3 * ENT_D + 3 * NENT);
        kzero<<<div_up(zn, 256), 256, 0, stream>>>(agg1s, zn);
        kedge_scatter<<<div_up((long)NEDGE * 16, 256), 256, 0, stream>>>(
            ei, et, drop, hop, ent_cur, relw, agg1s, agg2s, rels, cwc, cws, cew);
        kmean<<<div_up((long)ENT_D, 256), 256, 0, stream>>>(agg1s, cwc, (long)ENT_D);
        kmean<<<div_up((long)ENT_D, 256), 256, 0, stream>>>(agg2s, cws, (long)ENT_D);
        kmean<<<div_up((long)ENT_D, 256), 256, 0, stream>>>(rels,  cew, (long)ENT_D);
        kgemm_fused<<<(NENT + 127) / 128, 256, 0, stream>>>(
            agg1s, agg2s, W1w + hop * 4096, W1b + hop * 64,
            W2w + hop * 4096, W2b + hop * 64, ent_new, NENT);

        // item_agg = seg_mean(ucf_cur[mat_row], mat_col) -- uses hop-entry ucf
        long zi = (long)(ITM_D + NITM);
        kzero<<<div_up(zi, 256), 256, 0, stream>>>(icf_new, zi);
        kitem_scatter<<<div_up((long)NINTER * 16, 256), 256, 0, stream>>>(im, ucf_cur, icf_new, cnt_it);
        kmean<<<div_up((long)ITM_D, 256), 256, 0, stream>>>(icf_new, cnt_it, (long)ITM_D);

        // iterative attentive aggregation (ping-pong: cur -> tmp -> cur -> tmp)
        for (int it = 0; it < 3; ++it) {
            float* usrc = (it == 1) ? u_tmp   : usr_cur;
            float* qsrc = (it == 1) ? ucf_tmp : ucf_cur;
            float* udst = (it == 1) ? usr_cur : u_tmp;
            float* qdst = (it == 1) ? ucf_cur : ucf_tmp;

            kzero<<<div_up((long)(4 * NUSR), 256), 256, 0, stream>>>(segm_p, (long)(4 * NUSR));
            kdots<<<div_up((long)NINTER, 256), 256, 0, stream>>>(
                im, usrc, qsrc, rels, ent_cur, icf_cur, p_arr, q_arr, (int*)segm_p, (int*)segm_q);
            ksoftexp<<<div_up((long)NINTER, 256), 256, 0, stream>>>(
                im, p_arr, q_arr, (const int*)segm_p, segs_p, (const int*)segm_q, segs_q);
            knormmask<<<div_up((long)NINTER, 256), 256, 0, stream>>>(
                im, p_arr, q_arr, segs_p, segs_q, mask_f);

            kzero<<<div_up((long)(2 * USR_D), 256), 256, 0, stream>>>(udst, (long)(2 * USR_D));
            kscatter_u<<<div_up((long)NINTER * 16, 256), 256, 0, stream>>>(
                im, p_arr, q_arr, mask_f, ent_cur, icf_cur, udst, qdst);
            if (it < 2) {
                kl2norm<<<NUSR, 64, 0, stream>>>(udst, NUSR);
                kl2norm<<<NUSR, 64, 0, stream>>>(qdst, NUSR);
            }
        }

        // finalize hop: normalize and accumulate residuals; update current embeddings
        knorm_add<<<NENT, 64, 0, stream>>>(ent_new, ent_cur, out_ent, NENT);
        knorm_add<<<NUSR, 64, 0, stream>>>(u_tmp,   usr_cur, out_usr, NUSR);
        knorm_add<<<NUSR, 64, 0, stream>>>(ucf_tmp, ucf_cur, out_ucf, NUSR);
        knorm_add<<<NITM, 64, 0, stream>>>(icf_new, icf_cur, out_icf, NITM);
    }

    kbank<<<div_up((long)NINTER, 256), 256, 0, stream>>>(mask_f, out_bank);
}
